// VectorQuantizer_67267777790227
// MI455X (gfx1250) — compile-verified, matrix operand order checked by reading, not run on hardware
//
#include <hip/hip_runtime.h>
#include <hip/hip_bf16.h>

typedef __attribute__((ext_vector_type(16))) __bf16 v16bf;
typedef __attribute__((ext_vector_type(8)))  float  v8f;

#define N_ROWS  32768   // 8*4096
#define DIM     256
#define KCODES  2048
#define NTILES  128     // KCODES/16
#define DCHUNKS 8       // DIM/32

// ---------------------------------------------------------------------------
// Zero the histogram (must be re-zeroed every launch: deterministic replay).
__global__ __launch_bounds__(256) void vq_init_kernel(float* __restrict__ counts) {
    int i = blockIdx.x * 256 + threadIdx.x;
    if (i < KCODES) counts[i] = 0.0f;
}

// ---------------------------------------------------------------------------
// Repack codebook into WMMA B-operand order, split into bf16 hi + lo blobs.
// Blob index = ((tile*8 + chunk)*32 + lane), each entry = 16 bf16 (32 bytes).
// B-operand (bf16 32x16, lane = N, VGPRs over K):
//   lane<16 : K = base+0..7, base+16..23  (base = chunk*32)
//   lane>=16: K = base+8..15, base+24..31
__global__ __launch_bounds__(256)
void vq_pack_kernel(const float* __restrict__ emb,
                    v16bf* __restrict__ bhi, v16bf* __restrict__ blo) {
    const int gid    = blockIdx.x * 256 + threadIdx.x;   // 0 .. 32767
    const int lane   = gid & 31;
    const int chunk  = (gid >> 5) & 7;
    const int tile   = gid >> 8;
    const int laneHi = lane >> 4;
    const int n      = tile * 16 + (lane & 15);
    const int base   = chunk * 32 + laneHi * 8;
    const float* er  = emb + (size_t)n * DIM;
    v16bf h, l;
#pragma unroll
    for (int j = 0; j < 16; ++j) {
        const int d = base + (j < 8 ? j : j + 8);        // 0..7 then 16..23
        float  x  = er[d];
        __bf16 hb = (__bf16)x;
        h[j] = hb;
        l[j] = (__bf16)(x - (float)hb);
    }
    bhi[gid] = h;
    blo[gid] = l;
}

// ---------------------------------------------------------------------------
// ||e_k||^2 in full f32.
__global__ __launch_bounds__(256)
void vq_enorms_kernel(const float* __restrict__ emb, float* __restrict__ enorms) {
    const int n = blockIdx.x * 256 + threadIdx.x;
    if (n >= KCODES) return;
    const float4* e = (const float4*)(emb + (size_t)n * DIM);
    float s = 0.0f;
#pragma unroll 4
    for (int i = 0; i < DIM / 4; ++i) {
        float4 v = e[i];
        s += v.x * v.x + v.y * v.y + v.z * v.z + v.w * v.w;
    }
    enorms[n] = s;
}

// ---------------------------------------------------------------------------
// Fused GEMM + argmin. 4 waves/block, wave owns 16 rows; A (hi+lo bf16) lives
// in registers across the whole codebook sweep. bf16x3 split: hi*hi+hi*lo+lo*hi
// accumulated into one f32 WMMA accumulator. argmin over t = ||e||^2 - 2 z.e
// (monotone in the reference distance per row), first-index tie-break.
__global__ __launch_bounds__(128)
void vq_gemm_argmin_kernel(const float* __restrict__ z,
                           const v16bf* __restrict__ bhi,
                           const v16bf* __restrict__ blo,
                           const float* __restrict__ enorms,
                           int* __restrict__ idxout) {
    const int wave   = threadIdx.x >> 5;
    const int lane   = threadIdx.x & 31;
    const int laneHi = lane >> 4;
    const int lrow   = lane & 15;
    const int row0   = blockIdx.x * 64 + wave * 16;
    const int myrow  = row0 + lrow;          // A-operand: lanes p and p+16 share row

    // --- load + split A into registers (16 rows x 256 d, hi+lo) -------------
    v16bf ahi[DCHUNKS], alo[DCHUNKS];
    const float* zr = z + (size_t)myrow * DIM;
#pragma unroll
    for (int c = 0; c < DCHUNKS; ++c) {
        const int base = c * 32 + laneHi * 8;
        float4 f0 = *(const float4*)(zr + base);
        float4 f1 = *(const float4*)(zr + base + 4);
        float4 f2 = *(const float4*)(zr + base + 16);
        float4 f3 = *(const float4*)(zr + base + 20);
        float xs[16] = {f0.x, f0.y, f0.z, f0.w, f1.x, f1.y, f1.z, f1.w,
                        f2.x, f2.y, f2.z, f2.w, f3.x, f3.y, f3.z, f3.w};
        v16bf h, l;
#pragma unroll
        for (int j = 0; j < 16; ++j) {
            float  x  = xs[j];
            __bf16 hb = (__bf16)x;
            h[j] = hb;
            l[j] = (__bf16)(x - (float)hb);
        }
        ahi[c] = h;
        alo[c] = l;
    }

    float minv[8];
    int   mini[8];
#pragma unroll
    for (int g = 0; g < 8; ++g) { minv[g] = 3.4e38f; mini[g] = 0; }

    // --- sweep all 128 column tiles of the codebook -------------------------
    for (int tile = 0; tile < NTILES; ++tile) {
        const v16bf* bh_p = bhi + (size_t)tile * (DCHUNKS * 32) + lane;
        const v16bf* bl_p = blo + (size_t)tile * (DCHUNKS * 32) + lane;
        if (tile + 1 < NTILES) {               // uniform branch: EXEC stays full
            __builtin_prefetch(bh_p + DCHUNKS * 32, 0, 1);
            __builtin_prefetch(bl_p + DCHUNKS * 32, 0, 1);
        }
        v8f acc = {0.f, 0.f, 0.f, 0.f, 0.f, 0.f, 0.f, 0.f};
#pragma unroll
        for (int c = 0; c < DCHUNKS; ++c) {
            v16bf bh = bh_p[c * 32];
            v16bf bl = bl_p[c * 32];
            acc = __builtin_amdgcn_wmma_f32_16x16x32_bf16(false, ahi[c], false, bh,
                                                          (short)0, acc, false, false);
            acc = __builtin_amdgcn_wmma_f32_16x16x32_bf16(false, ahi[c], false, bl,
                                                          (short)0, acc, false, false);
            acc = __builtin_amdgcn_wmma_f32_16x16x32_bf16(false, alo[c], false, bh,
                                                          (short)0, acc, false, false);
        }
        // C/D layout: VGPR g, lane l -> row M = g + 8*(l>=16), col N = l&15
        const int   n  = tile * 16 + lrow;
        const float en = enorms[n];
#pragma unroll
        for (int g = 0; g < 8; ++g) {
            float t = en - 2.0f * acc[g];
            if (t < minv[g]) { minv[g] = t; mini[g] = n; }   // strict < keeps first idx
        }
    }

    // --- reduce over the 16 columns held per half-wave (first-index ties) ---
#pragma unroll
    for (int g = 0; g < 8; ++g) {
        float v = minv[g];
        int   i = mini[g];
#pragma unroll
        for (int m = 1; m <= 8; m <<= 1) {      // xor 1,2,4,8 stays within half
            float ov = __shfl_xor(v, m, 32);
            int   oi = __shfl_xor(i, m, 32);
            if (ov < v || (ov == v && oi < i)) { v = ov; i = oi; }
        }
        if (lrow == 0) idxout[row0 + laneHi * 8 + g] = i;
    }
}

// ---------------------------------------------------------------------------
// Gather z_q, write z_q_st and idx(float), exact f32 sum of (z_q - z_e)^2
// (per-block partials -> deterministic), exact-integer float histogram.
__global__ __launch_bounds__(256)
void vq_gather_kernel(const float* __restrict__ z,
                      const float* __restrict__ emb,
                      const int* __restrict__ idx,
                      float* __restrict__ out_zq,
                      float* __restrict__ out_idx,
                      float* __restrict__ counts,
                      float* __restrict__ blocksums) {
    __shared__ float red[256];
    const int gid = blockIdx.x * 256 + threadIdx.x;  // 0 .. 2097151 (float4 units)
    const int row = gid >> 6;                        // 64 float4 per row
    const int d4  = gid & 63;
    const int k   = idx[row];
    const float4 zv = ((const float4*)z)[gid];
    const float4 ev = ((const float4*)emb)[(size_t)k * 64 + d4];
    float4 o;
    o.x = zv.x + (ev.x - zv.x);   // z_e + stop_grad(z_q - z_e), as written
    o.y = zv.y + (ev.y - zv.y);
    o.z = zv.z + (ev.z - zv.z);
    o.w = zv.w + (ev.w - zv.w);
    ((float4*)out_zq)[gid] = o;
    const float dx = ev.x - zv.x, dy = ev.y - zv.y, dz = ev.z - zv.z, dw = ev.w - zv.w;
    float s = dx * dx + dy * dy + dz * dz + dw * dw;
    if (d4 == 0) {
        out_idx[row] = (float)k;
        atomicAdd(&counts[k], 1.0f);   // integer-valued: exact in any order
    }
    red[threadIdx.x] = s;
    __syncthreads();
    for (int off = 128; off > 0; off >>= 1) {
        if (threadIdx.x < off) red[threadIdx.x] += red[threadIdx.x + off];
        __syncthreads();
    }
    if (threadIdx.x == 0) blocksums[blockIdx.x] = red[0];
}

// ---------------------------------------------------------------------------
// Deterministic scalar reduce: loss = 1.25*mean_sq - 0.01*perplexity.
__global__ __launch_bounds__(256)
void vq_finalize_kernel(const float* __restrict__ blocksums,
                        const float* __restrict__ counts,
                        float* __restrict__ out_loss) {
    __shared__ float red[256];
    const int t = threadIdx.x;
    float s = 0.0f;
    for (int j = t; j < 8192; j += 256) s += blocksums[j];
    red[t] = s;
    __syncthreads();
    for (int off = 128; off > 0; off >>= 1) {
        if (t < off) red[t] += red[t + off];
        __syncthreads();
    }
    const float S = red[0];
    __syncthreads();
    float h = 0.0f;
    for (int j = t; j < KCODES; j += 256) {
        float p = counts[j] * (1.0f / (float)N_ROWS);
        h += p * logf(p + 1e-10f);
    }
    red[t] = h;
    __syncthreads();
    for (int off = 128; off > 0; off >>= 1) {
        if (t < off) red[t] += red[t + off];
        __syncthreads();
    }
    if (t == 0) {
        const float M    = S * (1.0f / (float)(N_ROWS * DIM));
        const float perp = expf(-red[0]);
        out_loss[0] = 1.25f * M - 0.01f * perp;   // (0.25+1)*mean - 0.01*perplexity
    }
}

// ---------------------------------------------------------------------------
extern "C" void kernel_launch(void* const* d_in, const int* in_sizes, int n_in,
                              void* d_out, int out_size, void* d_ws, size_t ws_size,
                              hipStream_t stream) {
    const float* z   = (const float*)d_in[0];   // [8,4096,256] f32
    const float* emb = (const float*)d_in[1];   // [2048,256]  f32

    char* ws = (char*)d_ws;
    v16bf* bhi      = (v16bf*)(ws);                                    // 1 MB
    v16bf* blo      = (v16bf*)(ws + (1u << 20));                       // 1 MB
    float* enorms   = (float*)(ws + (2u << 20));                       // 8 KB
    int*   idxbuf   = (int*)  (ws + (2u << 20) + 8192);                // 128 KB
    float* counts   = (float*)(ws + (2u << 20) + 8192 + 131072);       // 8 KB
    float* blocksum = (float*)(ws + (2u << 20) + 8192 + 131072 + 8192);// 32 KB

    float* out_zq   = (float*)d_out;            // [8,4096,256]
    float* out_loss = out_zq + 8388608;         // scalar
    float* out_idx  = out_loss + 1;             // [8,4096] as float

    hipLaunchKernelGGL(vq_init_kernel,       dim3(8),    dim3(256), 0, stream, counts);
    hipLaunchKernelGGL(vq_pack_kernel,       dim3(128),  dim3(256), 0, stream, emb, bhi, blo);
    hipLaunchKernelGGL(vq_enorms_kernel,     dim3(8),    dim3(256), 0, stream, emb, enorms);
    hipLaunchKernelGGL(vq_gemm_argmin_kernel,dim3(512),  dim3(128), 0, stream, z, bhi, blo, enorms, idxbuf);
    hipLaunchKernelGGL(vq_gather_kernel,     dim3(8192), dim3(256), 0, stream, z, emb, idxbuf,
                       out_zq, out_idx, counts, blocksum);
    hipLaunchKernelGGL(vq_finalize_kernel,   dim3(1),    dim3(256), 0, stream, blocksum, counts, out_loss);
}
